// ChannelSegment_24292335026341
// MI455X (gfx1250) — compile-verified, matrix operand order checked by reading, not run on hardware
//
#include <hip/hip_runtime.h>
#include <hip/hip_bf16.h>
#include <math.h>

#define DEVI __device__ __forceinline__

typedef __attribute__((ext_vector_type(16))) __bf16        v16bf;
typedef __attribute__((ext_vector_type(8)))  float         v8f;
typedef __attribute__((ext_vector_type(8)))  unsigned int  v8u;

#define N_CH    4
#define CW      512
#define NHEAD   8
#define HD      64
#define HD2     32
#define LSEQ    1024
#define NB      2
#define MROWS   (NB * LSEQ)      /* 2048 */
#define QKVN    (3 * CW)         /* 1536 */
#define EPSV    1e-6f
#define LAMINIT 0.2f

DEVI unsigned short f2bf(float f) {
  union { float f; unsigned int u; } v; v.f = f;
  unsigned int r = v.u + 0x7FFFu + ((v.u >> 16) & 1u);   // RNE
  return (unsigned short)(r >> 16);
}
DEVI unsigned int pack2(float a, float b) {
  return (unsigned int)f2bf(a) | ((unsigned int)f2bf(b) << 16);
}
DEVI float silu(float x) { return x / (1.0f + __expf(-x)); }

union BF16x16 { v8u u; v16bf v; };

// Load 16 bf16 (two contiguous 16-byte chunks) into a WMMA A/B operand.
DEVI v16bf load_tile16(const unsigned short* p0, const unsigned short* p1) {
  uint4 a = *(const uint4*)p0;
  uint4 b = *(const uint4*)p1;
  BF16x16 r; r.u = (v8u){a.x, a.y, a.z, a.w, b.x, b.y, b.z, b.w};
  return r.v;
}

DEVI v8f wmma_bf16(v16bf a, v16bf b, v8f c) {
  return __builtin_amdgcn_wmma_f32_16x16x32_bf16(
      /*neg_a=*/false, a, /*neg_b=*/false, b,
      /*c_mod=*/(short)0, c, /*reuse_a=*/false, /*reuse_b=*/false);
}

DEVI float redmax16(float v) {
  v = fmaxf(v, __shfl_xor(v, 1, 32));
  v = fmaxf(v, __shfl_xor(v, 2, 32));
  v = fmaxf(v, __shfl_xor(v, 4, 32));
  v = fmaxf(v, __shfl_xor(v, 8, 32));
  return v;
}
DEVI float redsum16(float v) {
  v += __shfl_xor(v, 1, 32);
  v += __shfl_xor(v, 2, 32);
  v += __shfl_xor(v, 4, 32);
  v += __shfl_xor(v, 8, 32);
  return v;
}
DEVI float redsum32(float v) {
  v = redsum16(v);
  v += __shfl_xor(v, 16, 32);
  return v;
}

// ---------------- CDNA5 async global->LDS copy (GLOBAL_LOAD_ASYNC_TO_LDS_B128)
#if __has_builtin(__builtin_amdgcn_global_load_async_to_lds_b128)
#define HAVE_ASYNC_LDS 1
typedef int v4i_vs __attribute__((__vector_size__(16)));             // matches builtin proto
typedef __attribute__((address_space(1))) v4i_vs GVec;               // global int4
typedef __attribute__((address_space(3))) v4i_vs LVec;               // LDS int4
DEVI void async_cp_b128(const void* g, void* l) {
  // generic global addr == AS1 numerically; generic LDS addr low 32 bits == LDS offset
  GVec* gp = (GVec*)(unsigned long long)(size_t)g;
  LVec* lp = (LVec*)(unsigned int)(size_t)l;
  __builtin_amdgcn_global_load_async_to_lds_b128(gp, lp, 0, 0);
}
DEVI void async_wait0() {
#if __has_builtin(__builtin_amdgcn_s_wait_asynccnt)
  __builtin_amdgcn_s_wait_asynccnt(0);
#else
  asm volatile("s_wait_asynccnt 0x0" ::: "memory");
#endif
}
#endif

// ---------------------------------------------------------------- conversions
__global__ void conv_x_k(const float* __restrict__ x, unsigned short* __restrict__ xbf) {
  size_t i = ((size_t)blockIdx.x * 256 + threadIdx.x) * 4;
  float4 v = *(const float4*)(x + i);
  uint2 o; o.x = pack2(v.x, v.y); o.y = pack2(v.z, v.w);
  *(uint2*)(xbf + i) = o;
}

// W: [c][K][N] f32  ->  WT: [c][N][K] bf16  (one-time, tiny)
__global__ void conv_wT_k(const float* __restrict__ W, unsigned short* __restrict__ WT,
                          int K, int N) {
  int n = blockIdx.x, c = blockIdx.y;
  const float* src = W + (size_t)c * K * N + n;
  unsigned short* dst = WT + ((size_t)c * N + n) * K;
  for (int k = threadIdx.x; k < K; k += blockDim.x)
    dst[k] = f2bf(src[(size_t)k * N]);
}

// ------------------------------------------------- fused QKV GEMM + SiLU + RMS
// One wave computes a 16(M) x 64(N) tile: 4 accumulators, 16 K-steps, 64 WMMAs.
__global__ void __launch_bounds__(32) qkv_gemm_k(
    const unsigned short* __restrict__ xbf,
    const unsigned short* __restrict__ WT,      // [c][1536][512] bf16 (transposed)
    const float* __restrict__ bqkv,
    const float* __restrict__ wq,
    const float* __restrict__ wk,
    unsigned short* __restrict__ q1o, unsigned short* __restrict__ q2o,
    unsigned short* __restrict__ k1o, unsigned short* __restrict__ k2o,
    unsigned short* __restrict__ vTo) {
  const int lid  = threadIdx.x & 31;
  const int ln   = lid & 15;
  const int g    = lid >> 4;
  const int koff = g * 8;
  const int n0 = blockIdx.x * 64;
  const int m0 = blockIdx.y * 16;
  const int c  = blockIdx.z;

  const unsigned short* arow = xbf + (size_t)(m0 + ln) * (N_CH * CW) + c * CW;
  const unsigned short* brow[4];
#pragma unroll
  for (int t = 0; t < 4; ++t)
    brow[t] = WT + ((size_t)c * QKVN + (n0 + t * 16 + ln)) * CW;

  v8f acc[4] = {};
  for (int kb = 0; kb < CW; kb += 32) {
    v16bf a = load_tile16(arow + kb + koff, arow + kb + koff + 16);
#pragma unroll
    for (int t = 0; t < 4; ++t) {
      v16bf b = load_tile16(brow[t] + kb + koff, brow[t] + kb + koff + 16);
      acc[t] = wmma_bf16(a, b, acc[t]);
    }
    if (kb + 32 < CW) __builtin_prefetch(arow + kb + 32 + koff, 0, 1);
  }

  __shared__ float sm[16][64];
#pragma unroll
  for (int t = 0; t < 4; ++t) {
    float bias = bqkv[c * QKVN + n0 + t * 16 + ln];
#pragma unroll
    for (int r = 0; r < 8; ++r)
      sm[r + g * 8][t * 16 + ln] = silu(acc[t][r] + bias);
  }
  __syncthreads();

  const int which = n0 / CW;                 // 0=q 1=k 2=v
  const int hd    = (n0 % CW) / HD;
  const int b     = m0 >> 10;
  const int l0    = m0 & (LSEQ - 1);

  if (which < 2) {
    // per-(row,half) RMS over 32 elems + weight, store bf16 [c][b][h][l][32]
    int row = ln, half = g;
    float ss = 0.f;
#pragma unroll
    for (int j = 0; j < 32; ++j) { float t = sm[row][half * 32 + j]; ss += t * t; }
    float scl = rsqrtf(ss * (1.0f / 32.0f) + EPSV);
    const float* wv = (which == 0 ? wq : wk) + c * HD2;
    size_t base = (((size_t)(c * NB + b) * NHEAD + hd) * LSEQ + (l0 + row)) * HD2;
    unsigned short* dst = (which == 0 ? (half == 0 ? q1o : q2o)
                                      : (half == 0 ? k1o : k2o)) + base;
    unsigned int up[16];
#pragma unroll
    for (int j = 0; j < 16; ++j)
      up[j] = pack2(sm[row][half * 32 + 2 * j]     * wv[2 * j]     * scl,
                    sm[row][half * 32 + 2 * j + 1] * wv[2 * j + 1] * scl);
    uint4* d4 = (uint4*)dst;
    d4[0] = make_uint4(up[0],  up[1],  up[2],  up[3]);
    d4[1] = make_uint4(up[4],  up[5],  up[6],  up[7]);
    d4[2] = make_uint4(up[8],  up[9],  up[10], up[11]);
    d4[3] = make_uint4(up[12], up[13], up[14], up[15]);
  } else {
    // V transposed: [c][b][h][d][l] bf16 (B-operand layout for P.V)
#pragma unroll
    for (int rep = 0; rep < 2; ++rep) {
      int d = lid + rep * 32;
      unsigned short* dst =
          vTo + (((size_t)(c * NB + b) * NHEAD + hd) * HD + d) * LSEQ + l0;
      unsigned int up[8];
#pragma unroll
      for (int j = 0; j < 8; ++j)
        up[j] = pack2(sm[2 * j][d], sm[2 * j + 1][d]);
      uint4* d4 = (uint4*)dst;
      d4[0] = make_uint4(up[0], up[1], up[2], up[3]);
      d4[1] = make_uint4(up[4], up[5], up[6], up[7]);
    }
  }
}

// ----------------------------------------- dual causal flash attention + RMS
// One wave per (c,b,h, 16-row q-tile). Two online softmaxes, diff combine.
// V blocks are staged global->LDS with CDNA5 async copies, overlapped with the
// score WMMAs + softmax; ASYNCcnt waited only right before the P.V WMMAs.
__global__ void __launch_bounds__(32) attn_k(
    const unsigned short* __restrict__ q1, const unsigned short* __restrict__ q2,
    const unsigned short* __restrict__ k1, const unsigned short* __restrict__ k2,
    const unsigned short* __restrict__ vT,
    const float* __restrict__ lq1, const float* __restrict__ lk1,
    const float* __restrict__ lq2, const float* __restrict__ lk2,
    const float* __restrict__ wh,
    unsigned short* __restrict__ aout) {
  const int lid  = threadIdx.x & 31;
  const int ln   = lid & 15;
  const int g    = lid >> 4;
  const int koff = g * 8;
  const int m0 = blockIdx.x * 16;
  const int h  = blockIdx.y;
  const int cb = blockIdx.z;            // c*2 + b
  const int c  = cb >> 1;
  const int b  = cb & 1;

  // lambda = exp(lq1.lk1) - exp(lq2.lk2) + 0.2
  float e1 = lq1[c * HD2 + lid] * lk1[c * HD2 + lid];
  float e2 = lq2[c * HD2 + lid] * lk2[c * HD2 + lid];
  e1 = redsum32(e1); e2 = redsum32(e2);
  const float lam = __expf(e1) - __expf(e2) + LAMINIT;

  const size_t qkBase = (size_t)(cb * NHEAD + h) * LSEQ * HD2;
  const unsigned short* q1r = q1 + qkBase + (size_t)(m0 + ln) * HD2;
  const unsigned short* q2r = q2 + qkBase + (size_t)(m0 + ln) * HD2;
  const unsigned short* k1b = k1 + qkBase;
  const unsigned short* k2b = k2 + qkBase;
  const unsigned short* vb  = vT + (size_t)(cb * NHEAD + h) * HD * LSEQ;

  v16bf aQ1 = load_tile16(q1r + koff, q1r + koff + 16);
  v16bf aQ2 = load_tile16(q2r + koff, q2r + koff + 16);

  v8f O1[4] = {};
  v8f O2[4] = {};
  float mx1[8], mx2[8], l1[8], l2[8];
#pragma unroll
  for (int r = 0; r < 8; ++r) { mx1[r] = mx2[r] = -1e30f; l1[r] = l2[r] = 0.f; }

  __shared__ unsigned short sP1[16 * 32];
  __shared__ unsigned short sP2[16 * 32];
#ifdef HAVE_ASYNC_LDS
  __shared__ unsigned short sV[HD * 32];     // 64 rows (d) x 32 keys, 4 KB
#endif
  const float SCL = 0.17677669529663687f;   // 1/sqrt(32)
  const v8f zc = {};

  for (int kb = 0; kb < m0 + 16; kb += 32) {
#ifdef HAVE_ASYNC_LDS
    // Kick off V block copy (64 x 32 bf16): 256 x 16B chunks, 8 per lane.
#pragma unroll
    for (int i = 0; i < 8; ++i) {
      int ch = i * 32 + lid;
      int d = ch >> 2, part = ch & 3;
      async_cp_b128(vb + (size_t)d * LSEQ + kb + part * 8, &sV[d * 32 + part * 8]);
    }
#endif
    float s1v[2][8], s2v[2][8];
#pragma unroll
    for (int t = 0; t < 2; ++t) {
      const unsigned short* kr1 = k1b + (size_t)(kb + t * 16 + ln) * HD2;
      const unsigned short* kr2 = k2b + (size_t)(kb + t * 16 + ln) * HD2;
      v16bf bK1 = load_tile16(kr1 + koff, kr1 + koff + 16);
      v16bf bK2 = load_tile16(kr2 + koff, kr2 + koff + 16);
      v8f s1 = wmma_bf16(aQ1, bK1, zc);
      v8f s2 = wmma_bf16(aQ2, bK2, zc);
      int key = kb + t * 16 + ln;
#pragma unroll
      for (int r = 0; r < 8; ++r) {
        int row = m0 + r + g * 8;
        s1v[t][r] = (key > row) ? -1e30f : s1[r] * SCL;
        s2v[t][r] = (key > row) ? -1e30f : s2[r] * SCL;
      }
    }
    float p1a[8], p1b[8], p2a[8], p2b[8], al1[8], al2[8];
#pragma unroll
    for (int r = 0; r < 8; ++r) {
      float rm = redmax16(fmaxf(s1v[0][r], s1v[1][r]));
      float mn = fmaxf(mx1[r], rm);
      al1[r] = __expf(mx1[r] - mn); mx1[r] = mn;
      p1a[r] = __expf(s1v[0][r] - mn);
      p1b[r] = __expf(s1v[1][r] - mn);
      l1[r] = l1[r] * al1[r] + redsum16(p1a[r] + p1b[r]);

      rm = redmax16(fmaxf(s2v[0][r], s2v[1][r]));
      mn = fmaxf(mx2[r], rm);
      al2[r] = __expf(mx2[r] - mn); mx2[r] = mn;
      p2a[r] = __expf(s2v[0][r] - mn);
      p2b[r] = __expf(s2v[1][r] - mn);
      l2[r] = l2[r] * al2[r] + redsum16(p2a[r] + p2b[r]);
    }
    // P: C-layout -> LDS (row-major 16x32) for A-operand reload
#pragma unroll
    for (int r = 0; r < 8; ++r) {
      int row = r + g * 8;
      sP1[row * 32 + ln]      = f2bf(p1a[r]);
      sP1[row * 32 + 16 + ln] = f2bf(p1b[r]);
      sP2[row * 32 + ln]      = f2bf(p2a[r]);
      sP2[row * 32 + 16 + ln] = f2bf(p2b[r]);
    }
#pragma unroll
    for (int dt = 0; dt < 4; ++dt)
#pragma unroll
      for (int r = 0; r < 8; ++r) { O1[dt][r] *= al1[r]; O2[dt][r] *= al2[r]; }
    __syncthreads();
    v16bf aP1 = load_tile16(&sP1[ln * 32 + koff], &sP1[ln * 32 + koff + 16]);
    v16bf aP2 = load_tile16(&sP2[ln * 32 + koff], &sP2[ln * 32 + koff + 16]);
#ifdef HAVE_ASYNC_LDS
    async_wait0();                      // V tile resident in LDS
#pragma unroll
    for (int dt = 0; dt < 4; ++dt) {
      const unsigned short* vr = &sV[(dt * 16 + ln) * 32];
      v16bf bV = load_tile16(vr + koff, vr + koff + 16);
      O1[dt] = wmma_bf16(aP1, bV, O1[dt]);
      O2[dt] = wmma_bf16(aP2, bV, O2[dt]);
    }
#else
#pragma unroll
    for (int dt = 0; dt < 4; ++dt) {
      const unsigned short* vr = vb + (size_t)(dt * 16 + ln) * LSEQ + kb;
      v16bf bV = load_tile16(vr + koff, vr + koff + 16);
      O1[dt] = wmma_bf16(aP1, bV, O1[dt]);
      O2[dt] = wmma_bf16(aP2, bV, O2[dt]);
    }
#endif
    __syncthreads();
  }

  // diff = O1/l1 - lam*O2/l2 ; RMS(wh) * 0.8 ; store bf16 [c][row][CW]
  __shared__ float sO[16][64];
#pragma unroll
  for (int r = 0; r < 8; ++r) {
    float i1 = 1.0f / l1[r];
    float i2 = 1.0f / l2[r];
#pragma unroll
    for (int dt = 0; dt < 4; ++dt)
      sO[r + g * 8][dt * 16 + ln] = O1[dt][r] * i1 - lam * (O2[dt][r] * i2);
  }
  __syncthreads();
  {
    int row = ln, half = g;
    float ss = 0.f;
#pragma unroll
    for (int j = 0; j < 32; ++j) { float t = sO[row][half * 32 + j]; ss += t * t; }
    ss += __shfl_xor(ss, 16, 32);
    float scl = rsqrtf(ss * (1.0f / 64.0f) + EPSV) * (1.0f - LAMINIT);
    const float* whp = wh + c * HD + half * 32;
    unsigned short* dst =
        aout + ((size_t)c * MROWS + (b * LSEQ + m0 + row)) * CW + h * HD + half * 32;
    unsigned int up[16];
#pragma unroll
    for (int j = 0; j < 16; ++j)
      up[j] = pack2(sO[row][half * 32 + 2 * j]     * whp[2 * j]     * scl,
                    sO[row][half * 32 + 2 * j + 1] * whp[2 * j + 1] * scl);
    uint4* d4 = (uint4*)dst;
    d4[0] = make_uint4(up[0],  up[1],  up[2],  up[3]);
    d4[1] = make_uint4(up[4],  up[5],  up[6],  up[7]);
    d4[2] = make_uint4(up[8],  up[9],  up[10], up[11]);
    d4[3] = make_uint4(up[12], up[13], up[14], up[15]);
  }
}

// ------------------------------------------------- output GEMM + SiLU (f32)
__global__ void __launch_bounds__(32) out_gemm_k(
    const unsigned short* __restrict__ abf,     // [c][2048][512] bf16
    const unsigned short* __restrict__ WT,      // [c][512][512] bf16 (transposed)
    const float* __restrict__ bout,
    float* __restrict__ ao2) {                  // [c][2048][512] f32
  const int lid  = threadIdx.x & 31;
  const int ln   = lid & 15;
  const int g    = lid >> 4;
  const int koff = g * 8;
  const int n0 = blockIdx.x * 64;
  const int m0 = blockIdx.y * 16;
  const int c  = blockIdx.z;

  const unsigned short* arow = abf + ((size_t)c * MROWS + m0 + ln) * CW;
  const unsigned short* brow[4];
#pragma unroll
  for (int t = 0; t < 4; ++t)
    brow[t] = WT + ((size_t)c * CW + n0 + t * 16 + ln) * CW;

  v8f acc[4] = {};
  for (int kb = 0; kb < CW; kb += 32) {
    v16bf a = load_tile16(arow + kb + koff, arow + kb + koff + 16);
#pragma unroll
    for (int t = 0; t < 4; ++t) {
      v16bf bb = load_tile16(brow[t] + kb + koff, brow[t] + kb + koff + 16);
      acc[t] = wmma_bf16(a, bb, acc[t]);
    }
    if (kb + 32 < CW) __builtin_prefetch(arow + kb + 32 + koff, 0, 1);
  }
#pragma unroll
  for (int t = 0; t < 4; ++t) {
    float bias = bout[c * CW + n0 + t * 16 + ln];
#pragma unroll
    for (int r = 0; r < 8; ++r)
      ao2[((size_t)c * MROWS + m0 + r + g * 8) * CW + n0 + t * 16 + ln] =
          silu(acc[t][r] + bias);
  }
}

// ------------------------- final RMS(wn) + residual + routing weight + pack
__global__ void __launch_bounds__(256) final_k(
    const float* __restrict__ x,
    const float* __restrict__ ao2,
    const float* __restrict__ wn,
    const float* __restrict__ rw,
    float* __restrict__ out) {
  const int tid = threadIdx.x;
  const int r = blockIdx.x;          // row in [0, 2048)
  const int c = blockIdx.y;
  const int b = r >> 10;
  const float* arow = ao2 + ((size_t)c * MROWS + r) * CW;
  __shared__ float red[256];
  float ss = 0.f;
  for (int j = tid; j < CW; j += 256) { float t = arow[j]; ss += t * t; }
  red[tid] = ss;
  __syncthreads();
  for (int s = 128; s > 0; s >>= 1) {
    if (tid < s) red[tid] += red[tid + s];
    __syncthreads();
  }
  float scl = rsqrtf(red[0] * (1.0f / 512.0f) + EPSV);
  float w = rw[b * N_CH + c];
  const float* xrow = x + (size_t)r * (N_CH * CW) + c * CW;
  float* orow = out + (size_t)r * (N_CH * CW) + c * CW;
  for (int j = tid; j < CW; j += 256)
    orow[j] = (xrow[j] + arow[j] * scl * wn[c * CW + j]) * w;
}

// -------------------------------------------------------------------- launch
extern "C" void kernel_launch(void* const* d_in, const int* in_sizes, int n_in,
                              void* d_out, int out_size, void* d_ws, size_t ws_size,
                              hipStream_t stream) {
  (void)in_sizes; (void)n_in; (void)out_size; (void)ws_size;
  const float* x    = (const float*)d_in[0];
  const float* rw   = (const float*)d_in[1];
  const float* Wqkv = (const float*)d_in[2];
  const float* bqkv = (const float*)d_in[3];
  const float* Wout = (const float*)d_in[4];
  const float* bout = (const float*)d_in[5];
  const float* lq1  = (const float*)d_in[6];
  const float* lk1  = (const float*)d_in[7];
  const float* lq2  = (const float*)d_in[8];
  const float* lk2  = (const float*)d_in[9];
  const float* wq   = (const float*)d_in[10];
  const float* wk   = (const float*)d_in[11];
  const float* wh   = (const float*)d_in[12];
  const float* wn   = (const float*)d_in[13];
  float* out = (float*)d_out;

  char* ws = (char*)d_ws;
  unsigned short* xbf   = (unsigned short*)(ws + 0);          //  8,388,608 B
  unsigned short* wqkvT = (unsigned short*)(ws + 8388608);    //  6,291,456 B
  unsigned short* woutT = (unsigned short*)(ws + 14680064);   //  2,097,152 B
  unsigned short* q1    = (unsigned short*)(ws + 16777216);   //  4,194,304 B
  unsigned short* q2    = (unsigned short*)(ws + 20971520);
  unsigned short* k1    = (unsigned short*)(ws + 25165824);
  unsigned short* k2    = (unsigned short*)(ws + 29360128);
  unsigned short* vT    = (unsigned short*)(ws + 33554432);   //  8,388,608 B
  unsigned short* aout  = (unsigned short*)(ws + 41943040);   //  8,388,608 B
  float*          ao2   = (float*)(ws + 50331648);            // 16,777,216 B

  conv_x_k<<<4096, 256, 0, stream>>>(x, xbf);
  conv_wT_k<<<dim3(QKVN, N_CH), 256, 0, stream>>>(Wqkv, wqkvT, CW, QKVN);
  conv_wT_k<<<dim3(CW, N_CH), 256, 0, stream>>>(Wout, woutT, CW, CW);

  qkv_gemm_k<<<dim3(QKVN / 64, MROWS / 16, N_CH), 32, 0, stream>>>(
      xbf, wqkvT, bqkv, wq, wk, q1, q2, k1, k2, vT);

  attn_k<<<dim3(LSEQ / 16, NHEAD, N_CH * NB), 32, 0, stream>>>(
      q1, q2, k1, k2, vT, lq1, lk1, lq2, lk2, wh, aout);

  out_gemm_k<<<dim3(CW / 64, MROWS / 16, N_CH), 32, 0, stream>>>(
      aout, woutT, bout, ao2);

  final_k<<<dim3(MROWS, N_CH), 256, 0, stream>>>(x, ao2, wn, rw, out);
}
